// HilbertMambaScattered_31207232373059
// MI455X (gfx1250) — matrix operand order repaired and checked by reading, not compiled, not run on hardware
//
#include <hip/hip_runtime.h>
#include <hip/hip_bf16.h>
#include <stdint.h>

// ---------------------------------------------------------------- constants
#define BATCH   8
#define IMG     512
#define GRID_N  64
#define SEQ     4096
#define DM      256
#define DI      512
#define DSTATE  16
#define DTRANK  16
#define NLAYERS 4
#define KCONV   4
#define NCLS    10
#define ROWS    (BATCH * SEQ)        // 32768 token rows

// ---------------------------------------------------------------- bf16 utils
typedef __attribute__((ext_vector_type(16))) __bf16 v16bf;
typedef __attribute__((ext_vector_type(8)))  float  v8f;

__device__ __forceinline__ uint16_t f2bf(float f) {
  uint32_t u = __float_as_uint(f);
  uint32_t r = u + 0x7FFFu + ((u >> 16) & 1u);   // round-to-nearest-even
  return (uint16_t)(r >> 16);
}
__device__ __forceinline__ float bf2f(uint16_t h) {
  return __uint_as_float(((uint32_t)h) << 16);
}
__device__ __forceinline__ float softplusf(float x) {
  return (x > 20.f) ? x : log1pf(__expf(x));
}
__device__ __forceinline__ float siluf(float x) {
  return x / (1.f + __expf(-x));
}

// ---------------------------------------------------------------- f32 -> bf16
__global__ void k_f32_to_bf16(const float* __restrict__ in,
                              uint16_t* __restrict__ out, int n) {
  int i = blockIdx.x * blockDim.x + threadIdx.x;
  if (i < n) out[i] = f2bf(in[i]);
}

// ---------------------------------------------------------------- im2col + Hilbert permute (bf16)
__global__ void k_im2col_hilbert(const float* __restrict__ x,
                                 const int* __restrict__ perm,
                                 uint16_t* __restrict__ P) {
  long long i = (long long)blockIdx.x * blockDim.x + threadIdx.x;
  const long long total = (long long)BATCH * SEQ * 192;
  if (i >= total) return;
  int j    = (int)(i % 192);
  int dseq = (int)((i / 192) % SEQ);
  int b    = (int)(i / (192LL * SEQ));
  int p  = perm[dseq];
  int gy = p >> 6, gx = p & 63;
  int c  = j >> 6;              // channel 0..2
  int r  = j & 63;
  int py = r >> 3, px = r & 7;
  float v = x[(((long long)b * 3 + c) * IMG + (gy * 8 + py)) * IMG + (gx * 8 + px)];
  P[i] = f2bf(v);               // row-major [b*SEQ+dseq][192]
}

// ---------------------------------------------------------------- RMSNorm -> bf16
__global__ void k_rmsnorm_bf16(const float* __restrict__ e,
                               const float* __restrict__ w,
                               uint16_t* __restrict__ u) {
  int row = blockIdx.x;              // one token row per block
  int c   = threadIdx.x;             // 256 threads
  float v = e[(size_t)row * DM + c];
  float s = v * v;
  #pragma unroll
  for (int off = 16; off; off >>= 1) s += __shfl_down(s, off);
  __shared__ float red[8];
  int lane = c & 31, wv = c >> 5;
  if (lane == 0) red[wv] = s;
  __syncthreads();
  float tot = 0.f;
  #pragma unroll
  for (int i = 0; i < 8; ++i) tot += red[i];
  float scale = rsqrtf(tot / (float)DM + 1e-5f);
  u[(size_t)row * DM + c] = f2bf(v * scale * w[c]);
}

// ---------------------------------------------------------------- bf16 WMMA GEMM
// C[M,N] = A[M,K] * W[N,K]^T (+epilogue).  8 waves/block, wave tile 16M x NT*16N,
// block tile 128M x NT*16N.  Requirements: M % 128 == 0, K % 32 == 0,
// N >= blockIdx.y*64 + NT*16 (NT = compile-time active 16-col tiles, 3 or 4).
// W slab (NT*16 rows x 32 K) staged in LDS per K-chunk, shared by all 8 waves.
// EPI: 0 = store, 1 = +bias, 3 = accumulate onto residual (R may alias C).
#define WS_PITCH 40   // bf16 elements per LDS row (80 B, padded to spread banks)

template <int EPI, int NT>
__global__ void k_gemm_bf16(const uint16_t* __restrict__ A, int lda,
                            const uint16_t* __restrict__ W, int ldw,
                            float* __restrict__ C, int ldc,
                            int M, int N, int K,
                            const float* __restrict__ bias,
                            const float* __restrict__ R, int ldr) {
  __shared__ uint16_t Ws[64 * WS_PITCH];
  const int tid  = threadIdx.x;
  const int wave = tid >> 5;
  const int lane = tid & 31;
  const int half = lane >> 4;
  const int l15  = lane & 15;
  const int m0   = (blockIdx.x * 8 + wave) * 16;
  const int n0   = blockIdx.y * 64;

  v8f acc[NT];
  #pragma unroll
  for (int t = 0; t < NT; ++t) {
    #pragma unroll
    for (int v = 0; v < 8; ++v) {
      if (EPI == 3)
        acc[t][v] = R[(size_t)(m0 + v + half * 8) * ldr + (n0 + t * 16 + l15)];
      else
        acc[t][v] = 0.f;
    }
  }

  const uint16_t* Arow = A + (size_t)(m0 + l15) * lda;
  // cooperative W staging: thread -> (row 0..63, 16-byte segment 0..3)
  const int wrow = tid >> 2;
  const int wseg = tid & 3;
  const bool wvalid = (NT == 4) ? true : (wrow < NT * 16);  // compile-time for NT==4
  const uint16_t* Wsrc = W + (size_t)(n0 + wrow) * ldw + wseg * 8;

  for (int kc = 0; kc < K; kc += 32) {
    uint4 wq = make_uint4(0u, 0u, 0u, 0u);
    if (wvalid) wq = *(const uint4*)(Wsrc + kc);

    // A fragment: ISA 16x32 bf16 layout -> two contiguous b128 loads per lane
    union { v16bf v; uint4 q[2]; } a;
    a.q[0] = *(const uint4*)(Arow + kc + half * 8);
    a.q[1] = *(const uint4*)(Arow + kc + 16 + half * 8);

    __syncthreads();                       // WAR: prior iteration done reading Ws
    *(uint4*)(Ws + wrow * WS_PITCH + wseg * 8) = wq;
    __syncthreads();                       // RAW: Ws populated

    #pragma unroll
    for (int t = 0; t < NT; ++t) {         // straight-line: no branches, no phis
      const uint16_t* src = Ws + (t * 16 + l15) * WS_PITCH + half * 8;
      union { v16bf v; uint4 q[2]; } b;
      b.q[0] = *(const uint4*)(src);
      b.q[1] = *(const uint4*)(src + 16);
      acc[t] = __builtin_amdgcn_wmma_f32_16x16x32_bf16(
          false, a.v, false, b.v, (short)0, acc[t], false, false);
    }
  }

  #pragma unroll
  for (int t = 0; t < NT; ++t) {
    #pragma unroll
    for (int v = 0; v < 8; ++v) {
      int row = m0 + v + half * 8;
      int col = n0 + t * 16 + l15;
      float val = acc[t][v];
      if (EPI == 1) val += bias[col];
      C[(size_t)row * ldc + col] = val;
    }
  }
}

// ---------------------------------------------------------------- depthwise causal conv + SiLU -> bf16
__global__ void k_conv_silu(const float* __restrict__ xz,    // [ROWS][1024], x part = cols 0..511
                            const float* __restrict__ cw,    // [DI][4]
                            const float* __restrict__ cb,    // [DI]
                            uint16_t* __restrict__ xc) {     // bf16 [ROWS][DI]
  long long i = (long long)blockIdx.x * blockDim.x + threadIdx.x;
  const long long total = (long long)ROWS * DI;
  if (i >= total) return;
  int d = (int)(i % DI);
  int l = (int)((i / DI) % SEQ);
  int b = (int)(i / ((long long)DI * SEQ));
  float acc = cb[d];
  #pragma unroll
  for (int j = 0; j < KCONV; ++j) {
    int ls = l - (KCONV - 1) + j;
    if (ls >= 0)
      acc = fmaf(cw[d * KCONV + j],
                 xz[((size_t)b * SEQ + ls) * (2 * DI) + d], acc);
  }
  xc[i] = f2bf(siluf(acc));
}

// ---------------------------------------------------------------- dt projection (K=16) + softplus
__global__ void k_dt(const float* __restrict__ xdbl,   // [ROWS][48]
                     const float* __restrict__ dtw,    // [DI][16]
                     const float* __restrict__ dtb,    // [DI]
                     float* __restrict__ dt) {         // [ROWS][DI]
  long long i = (long long)blockIdx.x * blockDim.x + threadIdx.x;
  const long long total = (long long)ROWS * DI;
  if (i >= total) return;
  int d = (int)(i % DI);
  size_t row = (size_t)(i / DI);
  float acc = dtb[d];
  #pragma unroll
  for (int k = 0; k < DTRANK; ++k)
    acc = fmaf(xdbl[row * 48 + k], dtw[d * DTRANK + k], acc);
  dt[i] = softplusf(acc);
}

// ---------------------------------------------------------------- selective scan (serial over L)
// 1 block per batch image, 512 threads = one channel each, h[16] in registers,
// per-step B/C vectors broadcast via LDS.
__global__ void k_scan(const float* __restrict__ dt,     // [ROWS][DI]
                       const uint16_t* __restrict__ xc,  // bf16 [ROWS][DI]
                       const float* __restrict__ xdbl,   // [ROWS][48]: [16:32)=B, [32:48)=C
                       const float* __restrict__ z,      // xz + DI, stride 2*DI
                       const float* __restrict__ A_log,  // [DI][16]
                       const float* __restrict__ Dsk,    // [DI]
                       uint16_t* __restrict__ y) {       // bf16 [ROWS][DI]
  const int b = blockIdx.x;
  const int d = threadIdx.x;
  __shared__ float Bs[DSTATE], Cs[DSTATE];
  float Ar[DSTATE], h[DSTATE];
  #pragma unroll
  for (int n = 0; n < DSTATE; ++n) {
    Ar[n] = -__expf(A_log[d * DSTATE + n]);
    h[n]  = 0.f;
  }
  const float dsk = Dsk[d];
  for (int l = 0; l < SEQ; ++l) {
    size_t row = (size_t)b * SEQ + l;
    __syncthreads();
    if (d < 2 * DSTATE) {
      float v = xdbl[row * 48 + DSTATE + d];
      if (d < DSTATE) Bs[d] = v; else Cs[d - DSTATE] = v;
    }
    __syncthreads();
    float dtv = dt[row * DI + d];
    float xv  = bf2f(xc[row * DI + d]);
    float zv  = z[row * (2 * DI) + d];
    float dx  = dtv * xv;
    float acc = 0.f;
    #pragma unroll
    for (int n = 0; n < DSTATE; ++n) {
      float dA = __expf(dtv * Ar[n]);
      h[n] = fmaf(h[n], dA, dx * Bs[n]);
      acc  = fmaf(h[n], Cs[n], acc);
    }
    acc = fmaf(xv, dsk, acc);
    acc *= siluf(zv);
    y[row * DI + d] = f2bf(acc);
  }
}

// ---------------------------------------------------------------- final norm + mean pool + head
__global__ void k_head(const float* __restrict__ emb,    // [ROWS][DM]
                       const float* __restrict__ nf,     // [DM]
                       const float* __restrict__ hw,     // [NCLS][DM]
                       const float* __restrict__ hb,     // [NCLS]
                       float* __restrict__ out) {        // [BATCH][NCLS]
  int b = blockIdx.x;
  int c = threadIdx.x;                   // 256
  int lane = c & 31, wv = c >> 5;
  __shared__ float red[8];
  __shared__ float pooled[DM];
  float wf = nf[c];
  float accp = 0.f;
  for (int l = 0; l < SEQ; ++l) {
    float v = emb[((size_t)b * SEQ + l) * DM + c];
    float s = v * v;
    #pragma unroll
    for (int off = 16; off; off >>= 1) s += __shfl_down(s, off);
    if (lane == 0) red[wv] = s;
    __syncthreads();
    float tot = 0.f;
    #pragma unroll
    for (int i = 0; i < 8; ++i) tot += red[i];
    accp = fmaf(v * rsqrtf(tot / (float)DM + 1e-5f), wf, accp);
    __syncthreads();                     // red reused next iteration
  }
  pooled[c] = accp / (float)SEQ;
  __syncthreads();
  if (c < NCLS) {
    float a = hb[c];
    for (int k = 0; k < DM; ++k) a = fmaf(pooled[k], hw[c * DM + k], a);
    out[b * NCLS + c] = a;
  }
}

// ---------------------------------------------------------------- host
static inline int ceil_div(long long a, int b) { return (int)((a + b - 1) / b); }

extern "C" void kernel_launch(void* const* d_in, const int* in_sizes, int n_in,
                              void* d_out, int out_size, void* d_ws, size_t ws_size,
                              hipStream_t stream) {
  const float* x         = (const float*)d_in[0];
  const int*   perm      = (const int*)d_in[1];
  const float* patch_w   = (const float*)d_in[2];
  const float* patch_b   = (const float*)d_in[3];
  const float* norm_w    = (const float*)d_in[4];
  const float* in_proj_w = (const float*)d_in[5];
  const float* conv_w    = (const float*)d_in[6];
  const float* conv_b    = (const float*)d_in[7];
  const float* x_proj_w  = (const float*)d_in[8];
  const float* dt_w      = (const float*)d_in[9];
  const float* dt_b      = (const float*)d_in[10];
  const float* A_log     = (const float*)d_in[11];
  const float* D_skip    = (const float*)d_in[12];
  const float* out_w     = (const float*)d_in[13];
  const float* norm_f    = (const float*)d_in[14];
  const float* head_w    = (const float*)d_in[15];
  const float* head_b    = (const float*)d_in[16];
  float* out = (float*)d_out;

  // -------- workspace layout (~311 MB). im2col P aliases the dt buffer
  // (disjoint lifetimes: P is consumed by the patch GEMM before layer 0).
  char* ws = (char*)d_ws;
  const size_t OFF_EMB = 0;                                   // f32 [ROWS][256]
  const size_t OFF_U   = OFF_EMB + (size_t)ROWS * DM * 4;     // bf16 [ROWS][256]
  const size_t OFF_XZ  = OFF_U   + (size_t)ROWS * DM * 2;     // f32 [ROWS][1024]
  const size_t OFF_XC  = OFF_XZ  + (size_t)ROWS * 2 * DI * 4; // bf16 [ROWS][512]
  const size_t OFF_XD  = OFF_XC  + (size_t)ROWS * DI * 2;     // f32 [ROWS][48]
  const size_t OFF_DT  = OFF_XD  + (size_t)ROWS * 48 * 4;     // f32 [ROWS][512] (alias: im2col bf16)
  const size_t OFF_Y   = OFF_DT  + (size_t)ROWS * DI * 4;     // bf16 [ROWS][512]
  const size_t OFF_WB  = OFF_Y   + (size_t)ROWS * DI * 2;     // bf16 weight scratch (<1MB)

  float*    emb  = (float*)(ws + OFF_EMB);
  uint16_t* ub   = (uint16_t*)(ws + OFF_U);
  float*    xz   = (float*)(ws + OFF_XZ);
  uint16_t* xc   = (uint16_t*)(ws + OFF_XC);
  float*    xdbl = (float*)(ws + OFF_XD);
  float*    dt   = (float*)(ws + OFF_DT);
  uint16_t* P    = (uint16_t*)(ws + OFF_DT);   // alias
  uint16_t* yb   = (uint16_t*)(ws + OFF_Y);

  uint16_t* patch_wb = (uint16_t*)(ws + OFF_WB + 0);       // 256*192
  uint16_t* inp_wb   = (uint16_t*)(ws + OFF_WB + 102400);  // 1024*256
  uint16_t* out_wb   = (uint16_t*)(ws + OFF_WB + 626688);  // 256*512
  uint16_t* xp_wb    = (uint16_t*)(ws + OFF_WB + 888832);  // 48*512

  const dim3 blk256(256);

  // -------- patch embed: im2col (Hilbert-permuted) + GEMM 32768x256x192 (+bias)
  k_f32_to_bf16<<<ceil_div(DM * 192, 256), blk256, 0, stream>>>(patch_w, patch_wb, DM * 192);
  k_im2col_hilbert<<<ceil_div((long long)ROWS * 192, 256), blk256, 0, stream>>>(x, perm, P);
  k_gemm_bf16<1, 4><<<dim3(ROWS / 128, DM / 64), blk256, 0, stream>>>(
      P, 192, patch_wb, 192, emb, DM, ROWS, DM, 192, patch_b, nullptr, 0);

  // -------- Mamba layers
  for (int l = 0; l < NLAYERS; ++l) {
    const float* nw  = norm_w    + (size_t)l * DM;
    const float* ipw = in_proj_w + (size_t)l * 2 * DI * DM;
    const float* cw  = conv_w    + (size_t)l * DI * KCONV;
    const float* cb  = conv_b    + (size_t)l * DI;
    const float* xpw = x_proj_w  + (size_t)l * 48 * DI;
    const float* dtw = dt_w      + (size_t)l * DI * DTRANK;
    const float* dtb = dt_b      + (size_t)l * DI;
    const float* al  = A_log     + (size_t)l * DI * DSTATE;
    const float* dsk = D_skip    + (size_t)l * DI;
    const float* ow  = out_w     + (size_t)l * DM * DI;

    k_f32_to_bf16<<<ceil_div(2 * DI * DM, 256), blk256, 0, stream>>>(ipw, inp_wb, 2 * DI * DM);
    k_f32_to_bf16<<<ceil_div(DM * DI, 256), blk256, 0, stream>>>(ow, out_wb, DM * DI);
    k_f32_to_bf16<<<ceil_div(48 * DI, 256), blk256, 0, stream>>>(xpw, xp_wb, 48 * DI);

    // RMSNorm -> bf16
    k_rmsnorm_bf16<<<ROWS, blk256, 0, stream>>>(emb, nw, ub);
    // in_proj: xz[32768,1024] = u @ W^T
    k_gemm_bf16<0, 4><<<dim3(ROWS / 128, (2 * DI) / 64), blk256, 0, stream>>>(
        ub, DM, inp_wb, DM, xz, 2 * DI, ROWS, 2 * DI, DM, nullptr, nullptr, 0);
    // depthwise causal conv + SiLU
    k_conv_silu<<<ceil_div((long long)ROWS * DI, 256), blk256, 0, stream>>>(xz, cw, cb, xc);
    // x_proj: x_dbl[32768,48] = xc @ W^T  (N = 48 = exactly 3 WMMA tiles)
    k_gemm_bf16<0, 3><<<dim3(ROWS / 128, 1), blk256, 0, stream>>>(
        xc, DI, xp_wb, DI, xdbl, 48, ROWS, 48, DI, nullptr, nullptr, 0);
    // dt = softplus(x_dbl[:, :16] @ dt_w^T + dt_b)
    k_dt<<<ceil_div((long long)ROWS * DI, 256), blk256, 0, stream>>>(xdbl, dtw, dtb, dt);
    // selective scan
    k_scan<<<BATCH, dim3(DI), 0, stream>>>(dt, xc, xdbl, xz + DI, al, dsk, yb);
    // out_proj with residual accumulate into emb
    k_gemm_bf16<3, 4><<<dim3(ROWS / 128, DM / 64), blk256, 0, stream>>>(
        yb, DI, out_wb, DI, emb, DM, ROWS, DM, DI, nullptr, emb, DM);
  }

  // -------- final RMSNorm + mean pool + classifier head
  k_head<<<BATCH, blk256, 0, stream>>>(emb, norm_f, head_w, head_b, out);
}
